// BILSTM_MH_ATTN_84335977824393
// MI455X (gfx1250) — matrix operand order repaired and checked by reading, not compiled, not run on hardware
//
#include <hip/hip_runtime.h>
#include <hip/hip_bf16.h>
#include <math.h>

#define BB   4096
#define TSEQ 128
#define DTAG 194
#define DCOM 20
#define BT   (BB * TSEQ)

typedef __attribute__((ext_vector_type(16))) __bf16 v16bf;
typedef __attribute__((ext_vector_type(8)))  __bf16 v8bf;
typedef __attribute__((ext_vector_type(8)))  float  v8f;
typedef __attribute__((ext_vector_type(4)))  float  v4f;
typedef __attribute__((ext_vector_type(2)))  float  v2f;

__device__ __forceinline__ float sigf(float x) { return 1.0f / (1.0f + __expf(-x)); }
__device__ __forceinline__ float tanhfast(float x) { return 1.0f - 2.0f / (__expf(2.0f * x) + 1.0f); }

__device__ __forceinline__ v8f wmma_bf16(v16bf a, v16bf b, v8f c) {
  return __builtin_amdgcn_wmma_f32_16x16x32_bf16(false, a, false, b, (short)0, c, false, false);
}

// ---- packed B-fragment load: per lane one contiguous 32B chunk -> 2x global_load_b128 ----
__device__ __forceinline__ v16bf load_frag(const __bf16* frags, int tile) {
  int lane = threadIdx.x & 31;
  return *(const v16bf*)(frags + ((size_t)tile * 32 + lane) * 16);
}

// ---- A fragment from contiguous bf16 row (16B-aligned), vectorized ----
__device__ __forceinline__ v16bf a_from_bf16(const __bf16* row, int k0) {
  int kh = ((threadIdx.x >> 4) & 1) * 8;
  v8bf lo = *(const v8bf*)(row + k0 + kh);
  v8bf hi = *(const v8bf*)(row + k0 + 16 + kh);
  return __builtin_shufflevector(lo, hi, 0, 1, 2, 3, 4, 5, 6, 7, 8, 9, 10, 11, 12, 13, 14, 15);
}

// ---- A fragment from f32 row, full tile (8B-aligned rows -> float2 loads) ----
__device__ __forceinline__ v16bf a_from_f32_fast(const float* row, int k0) {
  int kh = ((threadIdx.x >> 4) & 1) * 8;
  const float* p0 = row + k0 + kh;
  const float* p1 = row + k0 + 16 + kh;
  v16bf a;
#pragma unroll
  for (int i = 0; i < 8; i += 2) {
    v2f x0 = *(const v2f*)(p0 + i);
    v2f x1 = *(const v2f*)(p1 + i);
    a[i] = (__bf16)x0.x; a[i + 1] = (__bf16)x0.y;
    a[8 + i] = (__bf16)x1.x; a[8 + i + 1] = (__bf16)x1.y;
  }
  return a;
}

// ---- A fragment from f32 row, K-edge tile: branchless clamped-index loads ----
__device__ __forceinline__ v16bf a_from_f32_edge(const float* row, int k0, int K) {
  int kh = ((threadIdx.x >> 4) & 1) * 8;
  v16bf a;
#pragma unroll
  for (int i = 0; i < 8; ++i) {
    int k1 = k0 + kh + i, k2 = k0 + 16 + kh + i;
    float x1 = row[k1 < K ? k1 : 0];
    float x2 = row[k2 < K ? k2 : 0];
    a[i]     = (__bf16)(k1 < K ? x1 : 0.0f);
    a[8 + i] = (__bf16)(k2 < K ? x2 : 0.0f);
  }
  return a;
}

// ---------------- one-time weight packing into B-fragment layout (zero padded) ------------
__global__ void pack_wb_k(const float* W, int N, int K, int NT, void* frags_v) {
  __bf16* frags = (__bf16*)frags_v;
  int tile = blockIdx.x;          // tile = kt*NT + nt
  int kt = tile / NT, nt = tile % NT;
  int lane = threadIdx.x & 31;
  int n = nt * 16 + (lane & 15);
  int kh = (lane & 16) >> 1;
#pragma unroll
  for (int e = 0; e < 16; ++e) {
    int k = kt * 32 + kh + (e < 8 ? e : 8 + e);
    float v = (n < N && k < K) ? W[(size_t)n * K + k] : 0.0f;
    frags[((size_t)tile * 32 + lane) * 16 + e] = (__bf16)v;
  }
}

__global__ void pack_bias_k(const float* a, const float* b, int N, int Npad, float* out) {
  int n = blockIdx.x * blockDim.x + threadIdx.x;
  if (n >= Npad) return;
  float v = 0.0f;
  if (n < N) { v = a[n]; if (b != nullptr) v += b[n]; }
  out[n] = v;
}

// ---------------- Generic WMMA GEMM: OUT[M, NT*16] = act(A[M,K] @ W^T + bias) -------------
// Compile-time K/LDA fold away all edge-tile checks. ldo == NT*16 by construction.
// Grid must be exact: M divisible by 128 (8 waves x 16 rows per block).
template <int NT, int KT, int K, int LDA, bool ABF, int ACT, bool OBF>
__global__ void gemm_wmma_k(const void* Aptr, const __bf16* Wf, const float* Bp, void* OUT) {
  constexpr int LDO = NT * 16;
  __shared__ __align__(16) __bf16 sepb[OBF ? 8 * 16 * LDO : 8];
  __shared__ __align__(16) float  sepf[OBF ? 8 : 8 * 16 * LDO];

  int wv = threadIdx.x >> 5;
  int wave = blockIdx.x * 8 + wv;
  int m0 = wave * 16;
  int lane = threadIdx.x & 31;
  int l15 = lane & 15, hi8 = (lane >> 4) * 8;

  v8f acc[NT];
#pragma unroll
  for (int nt = 0; nt < NT; ++nt) {
    float bv = Bp[nt * 16 + l15];
#pragma unroll
    for (int r = 0; r < 8; ++r) acc[nt][r] = bv;
  }

  const float*  rowf = (const float*)Aptr + (size_t)(m0 + l15) * LDA;
  const __bf16* rowb = (const __bf16*)Aptr + (size_t)(m0 + l15) * LDA;

#pragma unroll
  for (int kt = 0; kt < KT; ++kt) {
    constexpr_unused:;
    int k0 = kt * 32;
    v16bf a;
    if (ABF)               a = a_from_bf16(rowb, k0);
    else if (k0 + 32 <= K) a = a_from_f32_fast(rowf, k0);
    else                   a = a_from_f32_edge(rowf, k0, K);
#pragma unroll
    for (int nt = 0; nt < NT; ++nt)
      acc[nt] = wmma_bf16(a, load_frag(Wf, kt * NT + nt), acc[nt]);
  }

  // ---- staged epilogue: C layout -> per-wave LDS -> coalesced row stores ----
  if (OBF) {
    __bf16* sh = sepb + wv * 16 * LDO;
#pragma unroll
    for (int nt = 0; nt < NT; ++nt)
#pragma unroll
      for (int r = 0; r < 8; ++r) {
        float v = acc[nt][r];
        if (ACT == 1) v = fmaxf(v, 0.0f);
        if (ACT == 2) v = sigf(v);
        sh[(r + hi8) * LDO + nt * 16 + l15] = (__bf16)v;
      }
    __syncthreads();
    int row = lane >> 1, half = lane & 1;
    const v8bf* src = (const v8bf*)(sh + row * LDO + half * (NT * 8));
    v8bf* dst = (v8bf*)((__bf16*)OUT + (size_t)(m0 + row) * LDO + half * (NT * 8));
#pragma unroll
    for (int i = 0; i < NT; ++i) dst[i] = src[i];
  } else {
    float* sh = sepf + wv * 16 * LDO;
#pragma unroll
    for (int nt = 0; nt < NT; ++nt)
#pragma unroll
      for (int r = 0; r < 8; ++r) {
        float v = acc[nt][r];
        if (ACT == 1) v = fmaxf(v, 0.0f);
        if (ACT == 2) v = sigf(v);
        sh[(r + hi8) * LDO + nt * 16 + l15] = v;
      }
    __syncthreads();
    int row = lane >> 1, half = lane & 1;
    const v4f* src = (const v4f*)(sh + row * LDO + half * (NT * 8));
    v4f* dst = (v4f*)((float*)OUT + (size_t)(m0 + row) * LDO + half * (NT * 8));
#pragma unroll
    for (int i = 0; i < NT * 2; ++i) dst[i] = src[i];
  }
}

// ---------------- Fused BiLSTM recurrence: gridDim.y selects direction --------------------
template <int KT>
__global__ void lstm_rec_k(const void* in_seq_v,
                           const __bf16* fih_f, const __bf16* fhh_f, const float* bp_f,
                           const __bf16* fih_b, const __bf16* fhh_b, const float* bp_b,
                           void* out_seq_v, float* hT_f, float* cT_f, float* hT_b, float* cT_b) {
  __shared__ __align__(16) __bf16 sh_h[4][16 * 32];
  const __bf16* in_seq  = (const __bf16*)in_seq_v;
  __bf16*       out_seq = (__bf16*)out_seq_v;
  const int Kin = KT * 32;
  int dir = blockIdx.y;
  const __bf16* fih = dir ? fih_b : fih_f;
  const __bf16* fhh = dir ? fhh_b : fhh_f;
  const float*  bp  = dir ? bp_b  : bp_f;
  float* hT = dir ? hT_b : hT_f;
  float* cT = dir ? cT_b : cT_f;
  int dircol = dir * 32;

  int wv = threadIdx.x >> 5;
  int lane = threadIdx.x & 31;
  int l15 = lane & 15, hi = lane >> 4, hi8 = hi * 8;
  int b0 = blockIdx.x * 64 + wv * 16;

  v16bf bih[KT][8];
#pragma unroll
  for (int kt = 0; kt < KT; ++kt)
#pragma unroll
    for (int nt = 0; nt < 8; ++nt) bih[kt][nt] = load_frag(fih, kt * 8 + nt);
  v16bf bhh[8];
#pragma unroll
  for (int nt = 0; nt < 8; ++nt) bhh[nt] = load_frag(fhh, nt);
  float bias[8];
#pragma unroll
  for (int nt = 0; nt < 8; ++nt) bias[nt] = bp[nt * 16 + l15];

  v16bf hfrag;
#pragma unroll
  for (int i = 0; i < 16; ++i) hfrag[i] = (__bf16)0.0f;
  float cst[2][8], hst[2][8];
#pragma unroll
  for (int j = 0; j < 2; ++j)
#pragma unroll
    for (int r = 0; r < 8; ++r) { cst[j][r] = 0.0f; hst[j][r] = 0.0f; }

  const __bf16* arow = in_seq + (size_t)(b0 + l15) * TSEQ * Kin;

  for (int t = 0; t < TSEQ; ++t) {
    int tt = dir ? (TSEQ - 1 - t) : t;
    // prefetch next timestep's input row (global_prefetch_b8)
    int tn = dir ? (tt > 0 ? tt - 1 : 0) : (tt < TSEQ - 1 ? tt + 1 : TSEQ - 1);
    __builtin_prefetch(arow + (size_t)tn * Kin, 0, 3);

    v8f acc[8];
#pragma unroll
    for (int nt = 0; nt < 8; ++nt)
#pragma unroll
      for (int r = 0; r < 8; ++r) acc[nt][r] = bias[nt];

    // gates += x_t @ w_ih^T
#pragma unroll
    for (int kt = 0; kt < KT; ++kt) {
      v16bf a = a_from_bf16(arow + (size_t)tt * Kin, kt * 32);
#pragma unroll
      for (int nt = 0; nt < 8; ++nt) acc[nt] = wmma_bf16(a, bih[kt][nt], acc[nt]);
    }
    // gates += h @ w_hh^T (K=32: one WMMA per tile)
#pragma unroll
    for (int nt = 0; nt < 8; ++nt) acc[nt] = wmma_bf16(hfrag, bhh[nt], acc[nt]);

    // i=tiles 0-1, f=2-3, g=4-5, o=6-7; c tiles align with i tiles elementwise
#pragma unroll
    for (int j = 0; j < 2; ++j) {
#pragma unroll
      for (int r = 0; r < 8; ++r) {
        float iv = sigf(acc[j][r]);
        float fv = sigf(acc[2 + j][r]);
        float gv = tanhfast(acc[4 + j][r]);
        float ov = sigf(acc[6 + j][r]);
        float cn = fv * cst[j][r] + iv * gv;
        cst[j][r] = cn;
        float hv = ov * tanhfast(cn);
        hst[j][r] = hv;
        sh_h[wv][(r + hi8) * 32 + j * 16 + l15] = (__bf16)hv;
      }
    }
    __syncthreads();
    // coalesced h store: lane covers row l15, cols hi*16..hi*16+15 (16B chunks)
    {
      const v8bf* src = (const v8bf*)&sh_h[wv][l15 * 32 + hi * 16];
      v8bf* dst = (v8bf*)(out_seq + ((size_t)(b0 + l15) * TSEQ + tt) * 64 + dircol + hi * 16);
      dst[0] = src[0];
      dst[1] = src[1];
    }
    // h back in A-fragment layout for next step
    hfrag = a_from_bf16(&sh_h[wv][l15 * 32], 0);
    __syncthreads();
  }

  if (hT != nullptr) {
#pragma unroll
    for (int j = 0; j < 2; ++j)
#pragma unroll
      for (int r = 0; r < 8; ++r) {
        size_t idx = (size_t)(b0 + r + hi8) * 32 + j * 16 + l15;
        hT[idx] = hst[j][r];
        cT[idx] = cst[j][r];
      }
  }
}

// ---------------- pack 4 attention-head w_ih/b_ih into [16,64]/[16] -----------------------
__global__ void pack_attn_k(const float* w1, const float* w2, const float* w3, const float* w4,
                            const float* b1, const float* b2, const float* b3, const float* b4,
                            float* SW, float* SB) {
  int n = threadIdx.x;
  if (n >= 16) return;
  const float* ws[4] = {w1, w2, w3, w4};
  const float* bs[4] = {b1, b2, b3, b4};
  int head = n >> 2, g = n & 3;
  for (int k = 0; k < 64; ++k) SW[n * 64 + k] = ws[head][g * 64 + k];
  SB[n] = bs[head][g];
}

// ---------------- scalar attention LSTM scan (hidden=1), one thread per batch -------------
__global__ void attn_scan_k(const float* hs, const float* cs, const float* dh_w,
                            const float* dh_b, const float* dc_w, const float* dc_b,
                            const float* w_hh, const float* b_hh, const float* preA,
                            int headcol, float* scores) {
  int b = blockIdx.x * blockDim.x + threadIdx.x;
  if (b >= BB) return;
  float h = dh_b[0], c = dc_b[0];
  for (int k = 0; k < 32; ++k) {
    h += hs[(size_t)b * 32 + k] * dh_w[k];
    c += cs[(size_t)b * 32 + k] * dc_w[k];
  }
  float wi = w_hh[0], wf = w_hh[1], wg = w_hh[2], wo = w_hh[3];
  float bi = b_hh[0], bf = b_hh[1], bg = b_hh[2], bo = b_hh[3];
  float* srow = scores + (size_t)b * TSEQ;
  const float* prow = preA + (size_t)b * TSEQ * 16 + headcol;
  for (int t = 0; t < TSEQ; ++t) {
    const float* p = prow + (size_t)t * 16;   // 16B-aligned quad
    float gi = p[0] + h * wi + bi;
    float gf = p[1] + h * wf + bf;
    float gg = p[2] + h * wg + bg;
    float go = p[3] + h * wo + bo;
    c = sigf(gf) * c + sigf(gi) * tanhfast(gg);
    h = sigf(go) * tanhfast(c);
    srow[t] = h;
  }
}

// ---------------- softmax over T + pooling -> bf16 feat [B,288] ---------------------------
__global__ void attn_pool_k(const float* scores, const void* h2_v, const float* x_com,
                            void* feat_v) {
  __shared__ float w[TSEQ];
  const __bf16* h2 = (const __bf16*)h2_v;
  __bf16* feat = (__bf16*)feat_v;
  int b = blockIdx.x, head = blockIdx.y;
  int tid = threadIdx.x;  // 64
  const float* s = scores + ((size_t)head * BB + b) * TSEQ;
  for (int t = tid; t < TSEQ; t += 64) w[t] = s[t];
  __syncthreads();
  if (tid == 0) {
    float mx = -1e30f;
    for (int t = 0; t < TSEQ; ++t) mx = fmaxf(mx, w[t]);
    float sum = 0.0f;
    for (int t = 0; t < TSEQ; ++t) { w[t] = __expf(w[t] - mx); sum += w[t]; }
    float inv = 1.0f / sum;
    for (int t = 0; t < TSEQ; ++t) w[t] *= inv;
  }
  __syncthreads();
  float acc = 0.0f;
  for (int t = 0; t < TSEQ; ++t)
    acc += w[t] * (float)h2[((size_t)b * TSEQ + t) * 64 + tid];
  feat[(size_t)b * 288 + head * 64 + tid] = (__bf16)acc;
  if (head == 0) {
    if (tid < DCOM) feat[(size_t)b * 288 + 256 + tid] = (__bf16)x_com[(size_t)b * DCOM + tid];
    else if (tid < 32) feat[(size_t)b * 288 + 256 + tid] = (__bf16)0.0f;  // pad 276..287
  }
}

extern "C" void kernel_launch(void* const* d_in, const int* in_sizes, int n_in,
                              void* d_out, int out_size, void* d_ws, size_t ws_size,
                              hipStream_t stream) {
  (void)in_sizes; (void)n_in; (void)out_size; (void)ws_size;

  const float* x_tag = (const float*)d_in[0];
  const float* x_com = (const float*)d_in[1];
  const float* pre_w = (const float*)d_in[2];
  const float* pre_b = (const float*)d_in[3];

  char* wsp = (char*)d_ws;
  size_t off = 0;
  auto alloc = [&](size_t bytes) -> void* {
    void* p = wsp + off;
    off = (off + bytes + 255) & ~(size_t)255;
    return p;
  };
  void*  buf_pre = alloc((size_t)BT * 32 * 2);   // bf16 [BT,32]
  void*  buf_h1  = alloc((size_t)BT * 64 * 2);   // bf16 [B,T,64]
  void*  buf_h2  = alloc((size_t)BT * 64 * 2);
  float* hn_f = (float*)alloc((size_t)BB * 32 * 4);
  float* cn_f = (float*)alloc((size_t)BB * 32 * 4);
  float* hn_b = (float*)alloc((size_t)BB * 32 * 4);
  float* cn_b = (float*)alloc((size_t)BB * 32 * 4);
  float* preA   = (float*)alloc((size_t)BT * 16 * 4);
  float* scores = (float*)alloc((size_t)4 * BB * TSEQ * 4);
  void*  featb  = alloc((size_t)BB * 288 * 2);   // bf16 [B,288]
  void*  y1     = alloc((size_t)BB * 128 * 2);   // bf16
  void*  y2     = alloc((size_t)BB * 128 * 2);
  float* SW = (float*)alloc(16 * 64 * 4);
  float* SB = (float*)alloc(16 * 4);
  auto fr = [&](int tiles) -> __bf16* { return (__bf16*)alloc((size_t)tiles * 512 * 2); };
  __bf16* F_pre   = fr(14);   // KT=7, NT=2
  __bf16* F_r0ihf = fr(8);  __bf16* F_r0hhf = fr(8);
  __bf16* F_r0ihb = fr(8);  __bf16* F_r0hhb = fr(8);
  __bf16* F_r1ihf = fr(16); __bf16* F_r1hhf = fr(8);
  __bf16* F_r1ihb = fr(16); __bf16* F_r1hhb = fr(8);
  __bf16* F_attn  = fr(2);
  __bf16* F_p1    = fr(72);   // KT=9, NT=8
  __bf16* F_p2    = fr(32);
  __bf16* F_p3    = fr(4);
  float* Bp_pre = (float*)alloc(32 * 4);
  float* Bp_r0f = (float*)alloc(128 * 4); float* Bp_r0b = (float*)alloc(128 * 4);
  float* Bp_r1f = (float*)alloc(128 * 4); float* Bp_r1b = (float*)alloc(128 * 4);
  float* Bp_at  = (float*)alloc(16 * 4);
  float* Bp_p1  = (float*)alloc(128 * 4);
  float* Bp_p2  = (float*)alloc(128 * 4);
  float* Bp_p3  = (float*)alloc(16 * 4);

  const float* P = nullptr;
  auto in = [&](int i) { return (const float*)d_in[i]; };

  // ---- one-time packing (tiny) ----
  pack_wb_k<<<14, 32, 0, stream>>>(pre_w, 32, DTAG, 2, F_pre);
  pack_wb_k<<<8, 32, 0, stream>>>(in(4), 128, 32, 8, F_r0ihf);
  pack_wb_k<<<8, 32, 0, stream>>>(in(5), 128, 32, 8, F_r0hhf);
  pack_wb_k<<<8, 32, 0, stream>>>(in(8), 128, 32, 8, F_r0ihb);
  pack_wb_k<<<8, 32, 0, stream>>>(in(9), 128, 32, 8, F_r0hhb);
  pack_wb_k<<<16, 32, 0, stream>>>(in(12), 128, 64, 8, F_r1ihf);
  pack_wb_k<<<8, 32, 0, stream>>>(in(13), 128, 32, 8, F_r1hhf);
  pack_wb_k<<<16, 32, 0, stream>>>(in(16), 128, 64, 8, F_r1ihb);
  pack_wb_k<<<8, 32, 0, stream>>>(in(17), 128, 32, 8, F_r1hhb);
  pack_wb_k<<<72, 32, 0, stream>>>(in(52), 128, 276, 8, F_p1);
  pack_wb_k<<<32, 32, 0, stream>>>(in(54), 128, 128, 8, F_p2);
  pack_wb_k<<<4, 32, 0, stream>>>(in(56), 16, 128, 1, F_p3);
  pack_attn_k<<<1, 64, 0, stream>>>(in(24), in(32), in(40), in(48),
                                    in(26), in(34), in(42), in(50), SW, SB);
  pack_wb_k<<<2, 32, 0, stream>>>(SW, 16, 64, 1, F_attn);
  pack_bias_k<<<1, 64, 0, stream>>>(pre_b, P, 32, 32, Bp_pre);
  pack_bias_k<<<1, 128, 0, stream>>>(in(6), in(7), 128, 128, Bp_r0f);
  pack_bias_k<<<1, 128, 0, stream>>>(in(10), in(11), 128, 128, Bp_r0b);
  pack_bias_k<<<1, 128, 0, stream>>>(in(14), in(15), 128, 128, Bp_r1f);
  pack_bias_k<<<1, 128, 0, stream>>>(in(18), in(19), 128, 128, Bp_r1b);
  pack_bias_k<<<1, 64, 0, stream>>>(SB, P, 16, 16, Bp_at);
  pack_bias_k<<<1, 128, 0, stream>>>(in(53), P, 128, 128, Bp_p1);
  pack_bias_k<<<1, 128, 0, stream>>>(in(55), P, 128, 128, Bp_p2);
  pack_bias_k<<<1, 64, 0, stream>>>(in(57), P, 16, 16, Bp_p3);

  const int TPB = 256;

  // 1) pre-projection GEMM: f32 A (K=194 edge-tiled at compile time) -> bf16 [BT,32]
  gemm_wmma_k<2, 7, DTAG, DTAG, false, 0, true><<<BT / 128, TPB, 0, stream>>>(
      x_tag, F_pre, Bp_pre, buf_pre);

  // 2) layer-0 BiLSTM (Kin=32), both directions in one launch
  lstm_rec_k<1><<<dim3(BB / 64, 2), 128, 0, stream>>>(
      buf_pre, F_r0ihf, F_r0hhf, Bp_r0f, F_r0ihb, F_r0hhb, Bp_r0b,
      buf_h1, nullptr, nullptr, nullptr, nullptr);

  // 3) layer-1 BiLSTM (Kin=64), capture final h/c
  lstm_rec_k<2><<<dim3(BB / 64, 2), 128, 0, stream>>>(
      buf_h1, F_r1ihf, F_r1hhf, Bp_r1f, F_r1ihb, F_r1hhb, Bp_r1b,
      buf_h2, hn_f, cn_f, hn_b, cn_b);

  // 4) stacked attention input projection: [BT,64] -> f32 [BT,16]
  gemm_wmma_k<1, 2, 64, 64, true, 0, false><<<BT / 128, TPB, 0, stream>>>(
      buf_h2, F_attn, Bp_at, preA);

  // 5) per-head scalar scans (heads 0,1 use bwd final state; 2,3 use fwd)
  for (int i = 0; i < 4; ++i) {
    int base = 20 + 8 * i;
    const float* hs = (i < 2) ? hn_b : hn_f;
    const float* cs = (i < 2) ? cn_b : cn_f;
    attn_scan_k<<<BB / TPB, TPB, 0, stream>>>(
        hs, cs, in(base + 0), in(base + 1), in(base + 2), in(base + 3),
        in(base + 5), in(base + 7), preA, i * 4, scores + (size_t)i * BB * TSEQ);
  }

  // 6) softmax + pooling + x_com concat -> bf16 feat [B,288] (padded)
  attn_pool_k<<<dim3(BB, 4, 1), 64, 0, stream>>>(scores, buf_h2, x_com, featb);

  // 7) MLP head (bf16 activations, f32 accumulate; final f32 out)
  gemm_wmma_k<8, 9, 288, 288, true, 1, true><<<BB / 128, TPB, 0, stream>>>(
      featb, F_p1, Bp_p1, y1);
  gemm_wmma_k<8, 4, 128, 128, true, 1, true><<<BB / 128, TPB, 0, stream>>>(
      y1, F_p2, Bp_p2, y2);
  gemm_wmma_k<1, 4, 128, 128, true, 2, false><<<BB / 128, TPB, 0, stream>>>(
      y2, F_p3, Bp_p3, d_out);
}
